// HardAssignmentTextEncoder_41077067219202
// MI455X (gfx1250) — compile-verified, matrix operand order checked by reading, not run on hardware
//
#include <hip/hip_runtime.h>
#include <hip/hip_bf16.h>
#include <math.h>

// Problem dims (fixed by the reference)
#define VOCAB_  32000
#define D_      256    // WORD_DIM
#define E_      1024   // EMBED
#define B_      128
#define L_      512
#define H_      256    // HID
#define M_TOT   (B_ * L_)   // 65536 tokens

typedef __attribute__((ext_vector_type(16))) _Float16 v16h;
typedef __attribute__((ext_vector_type(8)))  _Float16 v8h;
typedef __attribute__((ext_vector_type(8)))  float    v8f;
typedef __attribute__((ext_vector_type(4)))  float    v4f;

// ---------------------------------------------------------------------------
// WMMA fragment loaders for V_WMMA_F32_16X16X32_F16 (wave32).
// ISA 16-bit A-matrix 16x32 layout:
//   lanes 0-15  (row = lane):    halfs 0-7 = K 0..7,  halfs 8-15 = K 16..23
//   lanes 16-31 (row = lane-16): halfs 0-7 = K 8..15, halfs 8-15 = K 24..31
// B fragment assumed mirror layout with lane <-> N column (B stored as N x K
// row-major, i.e. rows of the (out,in) weight matrix).
// ---------------------------------------------------------------------------
__device__ __forceinline__ v16h load_frag_f16(const _Float16* base, int ld,
                                              int row0, int k0, int lane) {
  int row = row0 + (lane & 15);
  int kb  = k0 + ((lane >> 4) << 3);
  const _Float16* p = base + (size_t)row * ld + kb;
  v8h lo = *(const v8h*)(p);        // K kb..kb+7
  v8h hi = *(const v8h*)(p + 16);   // K kb+16..kb+23
  return __builtin_shufflevector(lo, hi, 0,1,2,3,4,5,6,7,8,9,10,11,12,13,14,15);
}

__device__ __forceinline__ v16h load_frag_from_f32(const float* base, int ld,
                                                   int row0, int k0, int lane) {
  int row = row0 + (lane & 15);
  int kb  = k0 + ((lane >> 4) << 3);
  const float* p = base + (size_t)row * ld + kb;
  v4f a0 = *(const v4f*)(p);
  v4f a1 = *(const v4f*)(p + 4);
  v4f a2 = *(const v4f*)(p + 16);
  v4f a3 = *(const v4f*)(p + 20);
  v16h r;
#pragma unroll
  for (int i = 0; i < 4; ++i) {
    r[i]      = (_Float16)a0[i];
    r[4 + i]  = (_Float16)a1[i];
    r[8 + i]  = (_Float16)a2[i];
    r[12 + i] = (_Float16)a3[i];
  }
  return r;
}

// ---------------------------------------------------------------------------
// Kernel 1: convert GEMM weights to f16 (row-major (out, in) kept as-is;
// each WMMA B column n is weight row n).
// ---------------------------------------------------------------------------
__global__ void prep_weights_kernel(const float* __restrict__ conv2_w,
                                    const float* __restrict__ g1_w,
                                    _Float16* __restrict__ w2h,
                                    _Float16* __restrict__ g1h) {
  int i = blockIdx.x * blockDim.x + threadIdx.x;
  if (i < E_ * E_) w2h[i] = (_Float16)conv2_w[i];
  if (i < H_ * E_) g1h[i] = (_Float16)g1_w[i];
}

// ---------------------------------------------------------------------------
// Kernel 2: embedding gather + grouped conv1d (groups=128, 2 in-ch, k=3,
// pad=1) + exact GELU. One block per token; 3 embedding rows staged in LDS.
// Output h in f16, row-major (token, 1024).
// ---------------------------------------------------------------------------
__global__ void embed_conv_gelu_kernel(const int* __restrict__ x,
                                       const float* __restrict__ W_emb,
                                       const float* __restrict__ conv1_w,
                                       const float* __restrict__ conv1_b,
                                       _Float16* __restrict__ h) {
  int t = blockIdx.x;          // token id
  int b = t >> 9;              // /512
  int l = t & (L_ - 1);
  int tid = threadIdx.x;       // 256 threads

  __shared__ float emb[3][D_];
#pragma unroll
  for (int dt = 0; dt < 3; ++dt) {
    int l2 = l - 1 + dt;
    float v = 0.f;
    if (l2 >= 0 && l2 < L_) {
      int tok = x[b * L_ + l2];
      v = W_emb[(size_t)tok * D_ + tid];
    }
    emb[dt][tid] = v;
  }
  __syncthreads();

#pragma unroll
  for (int j = 0; j < 4; ++j) {
    int o  = tid * 4 + j;           // output channel 0..1023
    int g  = o >> 3;                // group (8 out-ch per group)
    int c0 = g * 2;                 // 2 input channels per group
    const float* wv = conv1_w + o * 6;   // layout (O, I=2, K=3)
    float acc = conv1_b[o];
#pragma unroll
    for (int dt = 0; dt < 3; ++dt) {
      acc += emb[dt][c0] * wv[dt] + emb[dt][c0 + 1] * wv[3 + dt];
    }
    // exact (erf) GELU
    float ge = 0.5f * acc * (1.0f + erff(acc * 0.70710678118654752f));
    h[(size_t)t * E_ + o] = (_Float16)ge;
  }
}

// ---------------------------------------------------------------------------
// Kernel 3: x_conv = h @ conv2_w^T  (M=65536, N=1024, K=1024), f16 WMMA,
// f32 accumulate. Block = 8 waves; wave w owns a 16(M) x 64(N) strip:
// four 16x16 N-tiles sharing ONE A fragment per K-step (4x A reuse).
// Block tile = 16 x 512; gridDim.y = 2 covers N=1024.
// ---------------------------------------------------------------------------
__global__ void conv2_gemm_kernel(const _Float16* __restrict__ h,
                                  const _Float16* __restrict__ w2h,
                                  const float* __restrict__ conv2_b,
                                  float* __restrict__ x_conv) {
  int wave = threadIdx.x >> 5;
  int lane = threadIdx.x & 31;
  int m0 = blockIdx.x * 16;
  int n0 = blockIdx.y * 512 + wave * 64;

  v8f c0 = {}, c1 = {}, c2 = {}, c3 = {};
#pragma unroll 2
  for (int k0 = 0; k0 < E_; k0 += 32) {
    v16h a  = load_frag_f16(h, E_, m0, k0, lane);
    v16h b0 = load_frag_f16(w2h, E_, n0,      k0, lane);
    v16h b1 = load_frag_f16(w2h, E_, n0 + 16, k0, lane);
    v16h b2 = load_frag_f16(w2h, E_, n0 + 32, k0, lane);
    v16h b3 = load_frag_f16(w2h, E_, n0 + 48, k0, lane);
    // prefetch next K-block of the B stream (lowers to global_prefetch_b8)
    if (k0 + 32 < E_) {
      __builtin_prefetch(w2h + (size_t)(n0 + (lane & 15)) * E_ + k0 + 32, 0, 1);
    }
    c0 = __builtin_amdgcn_wmma_f32_16x16x32_f16(false, a, false, b0,
                                                (short)0, c0, false, false);
    c1 = __builtin_amdgcn_wmma_f32_16x16x32_f16(false, a, false, b1,
                                                (short)0, c1, false, false);
    c2 = __builtin_amdgcn_wmma_f32_16x16x32_f16(false, a, false, b2,
                                                (short)0, c2, false, false);
    c3 = __builtin_amdgcn_wmma_f32_16x16x32_f16(false, a, false, b3,
                                                (short)0, c3, false, false);
  }

  // C/D layout: lane = N column (lane&15), VGPR r -> M = r + 8*(lane>=16)
  int nc = lane & 15;
  int mb = m0 + ((lane >> 4) << 3);
  v8f acc[4] = {c0, c1, c2, c3};
#pragma unroll
  for (int t = 0; t < 4; ++t) {
    int n    = n0 + t * 16 + nc;
    float bs = conv2_b[n];
#pragma unroll
    for (int r = 0; r < 8; ++r) {
      x_conv[(size_t)(mb + r) * E_ + n] = acc[t][r] + bs;
    }
  }
}

// ---------------------------------------------------------------------------
// Kernel 4: gate GEMM (M=65536, N=256, K=1024) fused with
// sigmoid(. + g1_b) * g2_w and reduction over hidden -> scores (B*L).
// Block = 8 waves, each wave owns two 16-wide N tiles (8*32 = 256 = HID).
// ---------------------------------------------------------------------------
__global__ void gate_scores_kernel(const float* __restrict__ x_conv,
                                   const _Float16* __restrict__ g1h,
                                   const float* __restrict__ g1_b,
                                   const float* __restrict__ g2_w,
                                   const float* __restrict__ g2_b,
                                   float* __restrict__ scores) {
  int wave = threadIdx.x >> 5;
  int lane = threadIdx.x & 31;
  int m0 = blockIdx.x * 16;
  int n0 = wave * 32;

  __shared__ float sc[16];
  if (threadIdx.x < 16) sc[threadIdx.x] = 0.f;
  __syncthreads();

  v8f c0 = {};
  v8f c1 = {};
#pragma unroll 2
  for (int k0 = 0; k0 < E_; k0 += 32) {
    v16h a  = load_frag_from_f32(x_conv, E_, m0, k0, lane);
    v16h b0 = load_frag_f16(g1h, E_, n0,      k0, lane);
    v16h b1 = load_frag_f16(g1h, E_, n0 + 16, k0, lane);
    c0 = __builtin_amdgcn_wmma_f32_16x16x32_f16(false, a, false, b0,
                                                (short)0, c0, false, false);
    c1 = __builtin_amdgcn_wmma_f32_16x16x32_f16(false, a, false, b1,
                                                (short)0, c1, false, false);
  }

  int n_lo = n0 + (lane & 15);
  int n_hi = n_lo + 16;
  float b0v = g1_b[n_lo], b1v = g1_b[n_hi];
  float w0v = g2_w[n_lo], w1v = g2_w[n_hi];

  float part[8];
#pragma unroll
  for (int r = 0; r < 8; ++r) {
    float s0 = 1.f / (1.f + expf(-(c0[r] + b0v)));
    float s1 = 1.f / (1.f + expf(-(c1[r] + b1v)));
    part[r] = s0 * w0v + s1 * w1v;
  }
  // reduce over the 16 N-lanes within each half-wave (butterfly)
#pragma unroll
  for (int r = 0; r < 8; ++r) {
#pragma unroll
    for (int mask = 1; mask < 16; mask <<= 1)
      part[r] += __shfl_xor(part[r], mask, 32);
  }
  if ((lane & 15) == 0) {
    int mbase = (lane >> 4) << 3;   // 0 for lanes 0-15, 8 for lanes 16-31
#pragma unroll
    for (int r = 0; r < 8; ++r) atomicAdd(&sc[mbase + r], part[r]);
  }
  __syncthreads();
  if (threadIdx.x < 16) scores[m0 + threadIdx.x] = sc[threadIdx.x] + g2_b[0];
}

// ---------------------------------------------------------------------------
// Kernel 5: per-batch softmax over L, rank-based top-256 selection (ties
// broken by lower index, matching lax.top_k), weighted mean of selected
// x_conv rows, L2 normalize. One 512-thread block per batch row.
// ---------------------------------------------------------------------------
__global__ void capsule_kernel(const float* __restrict__ scores,
                               const float* __restrict__ x_conv,
                               float* __restrict__ out) {
  int b = blockIdx.x;
  int i = threadIdx.x;   // 512
  __shared__ float w[L_];
  __shared__ float red[L_];

  float s = scores[b * L_ + i];
  red[i] = s; __syncthreads();
  for (int off = 256; off > 0; off >>= 1) {
    if (i < off) red[i] = fmaxf(red[i], red[i + off]);
    __syncthreads();
  }
  float mx = red[0];
  __syncthreads();
  float e = expf(s - mx);
  red[i] = e; __syncthreads();
  for (int off = 256; off > 0; off >>= 1) {
    if (i < off) red[i] += red[i + off];
    __syncthreads();
  }
  float wi = e / red[0];
  __syncthreads();
  w[i] = wi;
  __syncthreads();

  // rank of weight i (strictly-greater, or equal with lower index, wins)
  int rank = 0;
  for (int j = 0; j < L_; ++j) {
    float wj = w[j];
    rank += (wj > wi) || (wj == wi && j < i);
  }
  __syncthreads();
  w[i] = (rank < 256) ? wi : 0.f;   // k = L/2 = 256
  __syncthreads();

  // cap[e] = (1/k) * sum_selected w_j * x_conv[b, j, e]; thread i covers e=i, i+512
  const float* xr = x_conv + (size_t)b * L_ * E_;
  float a0 = 0.f, a1 = 0.f;
  for (int j = 0; j < L_; ++j) {
    float mj = w[j];
    if (mj != 0.f) {
      const float* row = xr + (size_t)j * E_;
      a0 += mj * row[i];
      a1 += mj * row[i + 512];
    }
  }
  a0 *= (1.f / 256.f);
  a1 *= (1.f / 256.f);

  red[i] = a0 * a0 + a1 * a1; __syncthreads();
  for (int off = 256; off > 0; off >>= 1) {
    if (i < off) red[i] += red[i + off];
    __syncthreads();
  }
  float inv = 1.f / fmaxf(sqrtf(red[0]), 1e-12f);
  out[(size_t)b * E_ + i]       = a0 * inv;
  out[(size_t)b * E_ + i + 512] = a1 * inv;
}

// ---------------------------------------------------------------------------
// kernel_launch: inputs in setup_inputs() order:
// 0 x(int), 1 lengths(int, unused), 2 W_emb, 3 conv1_w, 4 conv1_b,
// 5 conv2_w, 6 conv2_b, 7 g1_w, 8 g1_b, 9 g2_w, 10 g2_b.
// Output: (B,1,1024) f32.
// ---------------------------------------------------------------------------
extern "C" void kernel_launch(void* const* d_in, const int* in_sizes, int n_in,
                              void* d_out, int out_size, void* d_ws, size_t ws_size,
                              hipStream_t stream) {
  const int*   x       = (const int*)d_in[0];
  const float* W_emb   = (const float*)d_in[2];
  const float* conv1_w = (const float*)d_in[3];
  const float* conv1_b = (const float*)d_in[4];
  const float* conv2_w = (const float*)d_in[5];
  const float* conv2_b = (const float*)d_in[6];
  const float* g1_w    = (const float*)d_in[7];
  const float* g1_b    = (const float*)d_in[8];
  const float* g2_w    = (const float*)d_in[9];
  const float* g2_b    = (const float*)d_in[10];
  float* out = (float*)d_out;

  // Workspace layout (~406 MB)
  char* ws = (char*)d_ws;
  size_t off = 0;
  _Float16* h_f16  = (_Float16*)(ws + off); off += (size_t)M_TOT * E_ * sizeof(_Float16);
  float*    x_conv = (float*)   (ws + off); off += (size_t)M_TOT * E_ * sizeof(float);
  _Float16* w2h    = (_Float16*)(ws + off); off += (size_t)E_ * E_ * sizeof(_Float16);
  _Float16* g1h    = (_Float16*)(ws + off); off += (size_t)H_ * E_ * sizeof(_Float16);
  float*    scores = (float*)   (ws + off); off += (size_t)M_TOT * sizeof(float);
  (void)ws_size; (void)in_sizes; (void)n_in; (void)out_size;

  prep_weights_kernel<<<(E_ * E_) / 256, 256, 0, stream>>>(conv2_w, g1_w, w2h, g1h);
  embed_conv_gelu_kernel<<<M_TOT, 256, 0, stream>>>(x, W_emb, conv1_w, conv1_b, h_f16);
  conv2_gemm_kernel<<<dim3(M_TOT / 16, E_ / 512), 256, 0, stream>>>(h_f16, w2h, conv2_b, x_conv);
  gate_scores_kernel<<<M_TOT / 16, 256, 0, stream>>>(x_conv, g1h, g1_b, g2_w, g2_b, scores);
  capsule_kernel<<<B_, 512, 0, stream>>>(scores, x_conv, out);
}
